// GATModel_55508157333848
// MI455X (gfx1250) — matrix operand order repaired and checked by reading, not compiled, not run on hardware
//
#include <hip/hip_runtime.h>

// ---------------------------------------------------------------------------
// GAT model for MI455X (gfx1250, wave32).  GEMMs via v_wmma_f32_16x16x32_f16,
// edge softmax/aggregation via float atomics, everything graph-capture safe.
// ---------------------------------------------------------------------------

typedef __attribute__((ext_vector_type(16))) _Float16 v16h;
typedef __attribute__((ext_vector_type(8)))  _Float16 v8h;
typedef __attribute__((ext_vector_type(8)))  float    v8f;

constexpr int N_NODES  = 50000;
constexpr int N_EDGES  = 800000;
constexpr int E_TOT    = N_EDGES + N_NODES;   // + self loops
constexpr int IN_DIM   = 128;
constexpr int HID      = 32;
constexpr int HEADS    = 4;
constexpr int F1       = HEADS * HID;         // 128
constexpr int N_GRAPHS = 64;
constexpr int N_CLASSES= 10;
constexpr int M2PAD    = 50048;               // 50000 rounded up to 64

// ---------------------------- small helpers --------------------------------

__device__ __forceinline__ void atomic_max_f32(float* addr, float val) {
  // IEEE trick: signed-int max for >=0, unsigned-int min for <0.
  if (val >= 0.0f) atomicMax((int*)addr, __float_as_int(val));
  else             atomicMin((unsigned int*)addr, __float_as_uint(val));
}

__device__ __forceinline__ void edge_sd(const int* __restrict__ ei, int e,
                                        int& s, int& d) {
  if (e < N_EDGES) { s = ei[e]; d = ei[N_EDGES + e]; }
  else             { s = e - N_EDGES; d = s; }
}

// A-fragment for v_wmma_f32_16x16x32_f16 (ISA 16-bit A 16x32 layout):
// lanes 0-15: halves 0..7 <- K base..base+7, halves 8..15 <- K 16+base..,
// base = 0 (lanes 0-15) or 8 (lanes 16-31).
__device__ __forceinline__ v16h load_a_frag(const _Float16* __restrict__ A,
                                            int lda, int row, int kbase,
                                            int lane) {
  const int base = (lane & 16) ? 8 : 0;
  const _Float16* p = A + (size_t)row * lda + kbase + base;
  v8h lo = *(const v8h*)(p);        // 16B aligned
  v8h hi = *(const v8h*)(p + 16);
  v16h a;
#pragma unroll
  for (int i = 0; i < 8; ++i) { a[i] = lo[i]; a[i + 8] = hi[i]; }
  return a;
}

// ------------------------------- kernels -----------------------------------

__global__ void k_fill_f32(float* __restrict__ p, float v, int n) {
  int t = blockIdx.x * blockDim.x + threadIdx.x;
  if (t < n) p[t] = v;
}

__global__ void k_cvt_f16(const float* __restrict__ in,
                          _Float16* __restrict__ out, int n) {
  int t = blockIdx.x * blockDim.x + threadIdx.x;
  if (t < n) out[t] = (_Float16)in[t];
}

// Pack weight matrix W[K=128][ncols] (row-major f32) into per-wave WMMA B
// fragments: out[((ks*NT + ntile)*32 + lane)*16 + h] , K = ks*32 + hi8 + h + (h>=8 ? 8 : 0),
// N = ntile*16 + (lane&15), hi8 = 8 if lane>=16.
__global__ void k_pack_b(const float* __restrict__ W, _Float16* __restrict__ out,
                         int ncols, int NT) {
  int t = blockIdx.x * blockDim.x + threadIdx.x;
  if (t >= 4 * NT * 32) return;
  const int lane  = t & 31;
  const int ntile = (t >> 5) % NT;
  const int ks    = t / (32 * NT);
  const int kb    = ks * 32 + ((lane & 16) ? 8 : 0);
  const int n     = ntile * 16 + (lane & 15);
  _Float16* dst = out + (size_t)t * 16;
#pragma unroll
  for (int h = 0; h < 16; ++h) dst[h] = (_Float16)W[(size_t)(kb + h + (h >= 8 ? 8 : 0)) * ncols + n];
}

// GEMM1: H1[50000,128] = Xh[50000,128] x W1p.  One 16-row stripe per block,
// 8 waves cover N=128.  4 chained WMMAs over K.
__global__ void __launch_bounds__(256) k_gemm1(const _Float16* __restrict__ Xh,
                                               const _Float16* __restrict__ Bp,
                                               float* __restrict__ H1) {
  const int lane  = threadIdx.x & 31;
  const int ntile = threadIdx.x >> 5;            // 0..7
  const int m0    = blockIdx.x << 4;
  const int arow  = m0 + (lane & 15);
  v8f c = {0.f, 0.f, 0.f, 0.f, 0.f, 0.f, 0.f, 0.f};
#pragma unroll
  for (int ks = 0; ks < 4; ++ks) {
    v16h a = load_a_frag(Xh, IN_DIM, arow, ks * 32, lane);
    v16h b = *(const v16h*)(Bp + ((size_t)((ks * 8 + ntile) * 32 + lane)) * 16);
    c = __builtin_amdgcn_wmma_f32_16x16x32_f16(false, a, false, b,
                                               (short)0, c, false, false);
  }
  const int col   = ntile * 16 + (lane & 15);
  const int rbase = m0 + ((lane & 16) ? 8 : 0);
#pragma unroll
  for (int r = 0; r < 8; ++r) H1[(size_t)(rbase + r) * F1 + col] = c[r];
}

// GEMM2: Z[50000,32] = H2h[M2PAD,128] x W2p.  8 waves = 4 (M) x 2 (N) tiles,
// block covers 64 rows x 32 cols.
__global__ void __launch_bounds__(256) k_gemm2(const _Float16* __restrict__ Ah,
                                               const _Float16* __restrict__ Bp,
                                               float* __restrict__ Z) {
  const int lane = threadIdx.x & 31;
  const int wave = threadIdx.x >> 5;
  const int mi = wave >> 1, ni = wave & 1;
  const int m0   = blockIdx.x * 64 + mi * 16;
  const int arow = m0 + (lane & 15);
  v8f c = {0.f, 0.f, 0.f, 0.f, 0.f, 0.f, 0.f, 0.f};
#pragma unroll
  for (int ks = 0; ks < 4; ++ks) {
    v16h a = load_a_frag(Ah, F1, arow, ks * 32, lane);
    v16h b = *(const v16h*)(Bp + ((size_t)((ks * 2 + ni) * 32 + lane)) * 16);
    c = __builtin_amdgcn_wmma_f32_16x16x32_f16(false, a, false, b,
                                               (short)0, c, false, false);
  }
  const int col   = ni * 16 + (lane & 15);
  const int rbase = m0 + ((lane & 16) ? 8 : 0);
#pragma unroll
  for (int r = 0; r < 8; ++r) {
    const int row = rbase + r;
    if (row < N_NODES) Z[(size_t)row * HID + col] = c[r];
  }
}

// Per-node attention logits: als[n,h] = sum_d H[n,h,d]*a_src[h,d] (same ald).
__global__ void k_node_alpha(const float* __restrict__ H,
                             const float* __restrict__ a_src,
                             const float* __restrict__ a_dst,
                             float* __restrict__ als, float* __restrict__ ald,
                             int Hh) {
  int t = blockIdx.x * blockDim.x + threadIdx.x;
  if (t >= N_NODES * Hh) return;
  const int n = t / Hh, h = t % Hh;
  const float* hp = H + ((size_t)n * Hh + h) * HID;
  const float* as = a_src + h * HID;
  const float* ad = a_dst + h * HID;
  float s0 = 0.f, s1 = 0.f;
#pragma unroll 8
  for (int d = 0; d < HID; ++d) { float v = hp[d]; s0 += v * as[d]; s1 += v * ad[d]; }
  als[t] = s0; ald[t] = s1;
}

__global__ void k_edge_max(const int* __restrict__ ei,
                           const float* __restrict__ als,
                           const float* __restrict__ ald,
                           float* __restrict__ amax, int Hh) {
  int t = blockIdx.x * blockDim.x + threadIdx.x;
  if (t >= E_TOT * Hh) return;
  const int e = t / Hh, h = t % Hh;
  int s, d; edge_sd(ei, e, s, d);
  float a = als[s * Hh + h] + ald[d * Hh + h];
  a = a > 0.f ? a : 0.2f * a;                    // leaky relu
  atomic_max_f32(&amax[d * Hh + h], a);
}

__global__ void k_fix_amax(float* __restrict__ amax, int n) {
  int t = blockIdx.x * blockDim.x + threadIdx.x;
  if (t >= n) return;
  float v = amax[t];
  if (!(v >= -3.0e38f)) amax[t] = 0.f;           // -inf / NaN -> 0
}

__global__ void k_edge_exp(const int* __restrict__ ei,
                           const float* __restrict__ als,
                           const float* __restrict__ ald,
                           const float* __restrict__ amax,
                           float* __restrict__ ebuf, float* __restrict__ den,
                           int Hh) {
  int t = blockIdx.x * blockDim.x + threadIdx.x;
  if (t >= E_TOT * Hh) return;
  const int e = t / Hh, h = t % Hh;
  int s, d; edge_sd(ei, e, s, d);
  float a = als[s * Hh + h] + ald[d * Hh + h];
  a = a > 0.f ? a : 0.2f * a;
  float ex = expf(a - amax[d * Hh + h]);
  ebuf[t] = ex;
  atomicAdd(&den[d * Hh + h], ex);
}

__global__ void k_edge_agg(const int* __restrict__ ei,
                           const float* __restrict__ ebuf,
                           const float* __restrict__ den,
                           const float* __restrict__ feat,
                           float* __restrict__ agg, int Hh) {
  int t = blockIdx.x * blockDim.x + threadIdx.x;
  if (t >= E_TOT * Hh * HID) return;
  const int c  = t & (HID - 1);
  const int eh = t >> 5;
  const int h  = eh % Hh;
  const int e  = eh / Hh;
  int s, d; edge_sd(ei, e, s, d);
  const float w = ebuf[eh] / (den[d * Hh + h] + 1e-16f);
  atomicAdd(&agg[((size_t)d * Hh + h) * HID + c],
            w * feat[((size_t)s * Hh + h) * HID + c]);
}

// y = ELU( (x + bias - m) * rsqrt(v+eps) * g + b ); optional f32 / f16 outs.
__global__ void k_bias_bn_elu(const float* __restrict__ X,
                              const float* __restrict__ bias,
                              const float* __restrict__ g,
                              const float* __restrict__ b,
                              const float* __restrict__ m,
                              const float* __restrict__ v,
                              float* __restrict__ Y, _Float16* __restrict__ Yh,
                              int C, int npad) {
  int t = blockIdx.x * blockDim.x + threadIdx.x;
  if (t >= npad * C) return;
  const int n = t / C, c = t % C;
  if (n >= N_NODES) { if (Yh) Yh[t] = (_Float16)0.f; return; }  // zero pad rows
  float x = X[t] + bias[c];
  x = (x - m[c]) * rsqrtf(v[c] + 1e-5f) * g[c] + b[c];
  x = x > 0.f ? x : (expf(x) - 1.f);
  if (Y)  Y[t]  = x;
  if (Yh) Yh[t] = (_Float16)x;
}

__global__ void k_pool(const float* __restrict__ H3, const int* __restrict__ batch,
                       float* __restrict__ sums, float* __restrict__ cnts) {
  int t = blockIdx.x * blockDim.x + threadIdx.x;
  if (t >= N_NODES * HID) return;
  const int n = t / HID, c = t % HID;
  const int gidx = batch[n];
  atomicAdd(&sums[gidx * HID + c], H3[t]);
  if (c == 0) atomicAdd(&cnts[gidx], 1.0f);
}

__global__ void k_head(const float* __restrict__ sums, const float* __restrict__ cnts,
                       const float* __restrict__ fc1w, const float* __restrict__ fc1b,
                       const float* __restrict__ fc2w, const float* __restrict__ fc2b,
                       float* __restrict__ out) {
  const int gidx = threadIdx.x;
  if (gidx >= N_GRAPHS) return;
  float pooled[HID], z[HID];
  const float inv = 1.0f / fmaxf(cnts[gidx], 1.0f);
#pragma unroll
  for (int c = 0; c < HID; ++c) pooled[c] = sums[gidx * HID + c] * inv;
  for (int j = 0; j < HID; ++j) {
    float s = fc1b[j];
    for (int c = 0; c < HID; ++c) s += pooled[c] * fc1w[c * HID + j];
    z[j] = s > 0.f ? s : (expf(s) - 1.f);
  }
  for (int k = 0; k < N_CLASSES; ++k) {
    float s = fc2b[k];
    for (int j = 0; j < HID; ++j) s += z[j] * fc2w[j * N_CLASSES + k];
    out[gidx * N_CLASSES + k] = s;
  }
}

// ------------------------------- launcher ----------------------------------

static inline int cdiv(long long a, long long b) { return (int)((a + b - 1) / b); }

extern "C" void kernel_launch(void* const* d_in, const int* in_sizes, int n_in,
                              void* d_out, int out_size, void* d_ws, size_t ws_size,
                              hipStream_t stream) {
  (void)in_sizes; (void)n_in; (void)out_size; (void)ws_size;
  const float* x      = (const float*)d_in[0];
  const int*   ei     = (const int*)d_in[1];
  const int*   batch  = (const int*)d_in[2];
  const float* W1     = (const float*)d_in[3];
  const float* a_src1 = (const float*)d_in[4];
  const float* a_dst1 = (const float*)d_in[5];
  const float* b1     = (const float*)d_in[6];
  const float* bn1_g  = (const float*)d_in[7];
  const float* bn1_b  = (const float*)d_in[8];
  const float* bn1_m  = (const float*)d_in[9];
  const float* bn1_v  = (const float*)d_in[10];
  const float* W2     = (const float*)d_in[11];
  const float* a_src2 = (const float*)d_in[12];
  const float* a_dst2 = (const float*)d_in[13];
  const float* b2     = (const float*)d_in[14];
  const float* bn2_g  = (const float*)d_in[15];
  const float* bn2_b  = (const float*)d_in[16];
  const float* bn2_m  = (const float*)d_in[17];
  const float* bn2_v  = (const float*)d_in[18];
  const float* fc1_w  = (const float*)d_in[19];
  const float* fc1_b  = (const float*)d_in[20];
  const float* fc2_w  = (const float*)d_in[21];
  const float* fc2_b  = (const float*)d_in[22];
  float* out = (float*)d_out;

  // bump allocator over workspace
  char* wsb = (char*)d_ws;
  size_t off = 0;
  auto alloc = [&](size_t bytes) -> void* {
    void* p = wsb + off;
    off = (off + bytes + 255) & ~(size_t)255;
    return p;
  };
  _Float16* xh   = (_Float16*)alloc((size_t)N_NODES * IN_DIM * 2);
  _Float16* w1p  = (_Float16*)alloc((size_t)4 * 8 * 32 * 16 * 2);
  _Float16* w2p  = (_Float16*)alloc((size_t)4 * 2 * 32 * 16 * 2);
  float* h1      = (float*)alloc((size_t)N_NODES * F1 * 4);
  float* als1    = (float*)alloc((size_t)N_NODES * HEADS * 4);
  float* ald1    = (float*)alloc((size_t)N_NODES * HEADS * 4);
  float* amax1   = (float*)alloc((size_t)N_NODES * HEADS * 4);
  float* den1    = (float*)alloc((size_t)N_NODES * HEADS * 4);
  float* e1      = (float*)alloc((size_t)E_TOT * HEADS * 4);
  float* agg1    = (float*)alloc((size_t)N_NODES * F1 * 4);
  _Float16* h2h  = (_Float16*)alloc((size_t)M2PAD * F1 * 2);
  float* z2      = (float*)alloc((size_t)N_NODES * HID * 4);
  float* als2    = (float*)alloc((size_t)N_NODES * 4);
  float* ald2    = (float*)alloc((size_t)N_NODES * 4);
  float* amax2   = (float*)alloc((size_t)N_NODES * 4);
  float* den2    = (float*)alloc((size_t)N_NODES * 4);
  float* e2      = (float*)alloc((size_t)E_TOT * 4);
  float* agg2    = (float*)alloc((size_t)N_NODES * HID * 4);
  float* h3      = (float*)alloc((size_t)N_NODES * HID * 4);
  float* sums    = (float*)alloc((size_t)N_GRAPHS * HID * 4);
  float* cnts    = (float*)alloc((size_t)N_GRAPHS * 4);

  const int B = 256;
  const float NEG_INF = -__builtin_huge_valf();

  // ---- layer 1 ----
  k_cvt_f16<<<cdiv((long long)N_NODES * IN_DIM, B), B, 0, stream>>>(x, xh, N_NODES * IN_DIM);
  k_pack_b<<<cdiv(4 * 8 * 32, B), B, 0, stream>>>(W1, w1p, F1, 8);
  k_pack_b<<<cdiv(4 * 2 * 32, B), B, 0, stream>>>(W2, w2p, HID, 2);

  k_gemm1<<<N_NODES / 16, 256, 0, stream>>>(xh, w1p, h1);
  k_node_alpha<<<cdiv(N_NODES * HEADS, B), B, 0, stream>>>(h1, a_src1, a_dst1, als1, ald1, HEADS);

  k_fill_f32<<<cdiv(N_NODES * HEADS, B), B, 0, stream>>>(amax1, NEG_INF, N_NODES * HEADS);
  k_fill_f32<<<cdiv(N_NODES * HEADS, B), B, 0, stream>>>(den1, 0.f, N_NODES * HEADS);
  k_fill_f32<<<cdiv((long long)N_NODES * F1, B), B, 0, stream>>>(agg1, 0.f, N_NODES * F1);

  k_edge_max<<<cdiv((long long)E_TOT * HEADS, B), B, 0, stream>>>(ei, als1, ald1, amax1, HEADS);
  k_fix_amax<<<cdiv(N_NODES * HEADS, B), B, 0, stream>>>(amax1, N_NODES * HEADS);
  k_edge_exp<<<cdiv((long long)E_TOT * HEADS, B), B, 0, stream>>>(ei, als1, ald1, amax1, e1, den1, HEADS);
  k_edge_agg<<<cdiv((long long)E_TOT * HEADS * HID, B), B, 0, stream>>>(ei, e1, den1, h1, agg1, HEADS);

  k_bias_bn_elu<<<cdiv((long long)M2PAD * F1, B), B, 0, stream>>>(
      agg1, b1, bn1_g, bn1_b, bn1_m, bn1_v, nullptr, h2h, F1, M2PAD);

  // ---- layer 2 ----
  k_gemm2<<<M2PAD / 64, 256, 0, stream>>>(h2h, w2p, z2);
  k_node_alpha<<<cdiv(N_NODES, B), B, 0, stream>>>(z2, a_src2, a_dst2, als2, ald2, 1);

  k_fill_f32<<<cdiv(N_NODES, B), B, 0, stream>>>(amax2, NEG_INF, N_NODES);
  k_fill_f32<<<cdiv(N_NODES, B), B, 0, stream>>>(den2, 0.f, N_NODES);
  k_fill_f32<<<cdiv((long long)N_NODES * HID, B), B, 0, stream>>>(agg2, 0.f, N_NODES * HID);

  k_edge_max<<<cdiv((long long)E_TOT, B), B, 0, stream>>>(ei, als2, ald2, amax2, 1);
  k_fix_amax<<<cdiv(N_NODES, B), B, 0, stream>>>(amax2, N_NODES);
  k_edge_exp<<<cdiv((long long)E_TOT, B), B, 0, stream>>>(ei, als2, ald2, amax2, e2, den2, 1);
  k_edge_agg<<<cdiv((long long)E_TOT * HID, B), B, 0, stream>>>(ei, e2, den2, z2, agg2, 1);

  k_bias_bn_elu<<<cdiv((long long)N_NODES * HID, B), B, 0, stream>>>(
      agg2, b2, bn2_g, bn2_b, bn2_m, bn2_v, h3, nullptr, HID, N_NODES);

  // ---- pool + head ----
  k_fill_f32<<<cdiv(N_GRAPHS * HID, B), B, 0, stream>>>(sums, 0.f, N_GRAPHS * HID);
  k_fill_f32<<<1, B, 0, stream>>>(cnts, 0.f, N_GRAPHS);
  k_pool<<<cdiv((long long)N_NODES * HID, B), B, 0, stream>>>(h3, batch, sums, cnts);
  k_head<<<1, 64, 0, stream>>>(sums, cnts, fc1_w, fc1_b, fc2_w, fc2_b, out);
}
